// Test_5050881540773
// MI455X (gfx1250) — compile-verified, matrix operand order checked by reading, not run on hardware
//
#include <hip/hip_runtime.h>

typedef __attribute__((ext_vector_type(16))) _Float16 v16h;
typedef __attribute__((ext_vector_type(8)))  float    v8f;
typedef __attribute__((ext_vector_type(4)))  float    f32x4;   // native vec4 (NT-builtin compatible)

// ---------------------------------------------------------------------------
// Kernel 0: zero the 2-word scale workspace (graph-capture safe, deterministic)
// ---------------------------------------------------------------------------
__global__ void qlin_ws_init(unsigned* ws) {
    if (threadIdx.x == 0) { ws[0] = 0u; ws[1] = 0u; }
}

// ---------------------------------------------------------------------------
// Kernel 1: abs-max reduction over x (float4 grid-stride) + weight abs-max.
// |x| >= 0 so uint-bit atomicMax preserves float ordering. Default (RT)
// temporal policy: this pass pulls the 48MB of x into the 192MB L2 so the
// GEMM pass re-reads it at L2 bandwidth instead of HBM.
// ---------------------------------------------------------------------------
__global__ void __launch_bounds__(256)
qlin_absmax(const f32x4* __restrict__ x4, int n4,
            const float* __restrict__ w, unsigned* __restrict__ ws) {
    __shared__ float red[256];
    float m = 0.0f;
    for (int i = blockIdx.x * blockDim.x + threadIdx.x; i < n4;
         i += gridDim.x * blockDim.x) {
        f32x4 v = x4[i];
        m = fmaxf(m, fmaxf(fmaxf(fabsf(v.x), fabsf(v.y)),
                           fmaxf(fabsf(v.z), fabsf(v.w))));
    }
    red[threadIdx.x] = m;
    __syncthreads();
    for (int s = 128; s > 0; s >>= 1) {
        if ((int)threadIdx.x < s)
            red[threadIdx.x] = fmaxf(red[threadIdx.x], red[threadIdx.x + s]);
        __syncthreads();
    }
    if (threadIdx.x == 0)
        atomicMax(&ws[0], __float_as_uint(red[0]));
    if (blockIdx.x == 0 && threadIdx.x == 0) {
        float wm = 0.0f;
        for (int i = 0; i < 9; ++i) wm = fmaxf(wm, fabsf(w[i]));
        ws[1] = __float_as_uint(wm);
    }
}

// ---------------------------------------------------------------------------
// Kernel 2: quantize + GEMM via v_wmma_f32_16x16x32_f16.
// Wave tile = 128 rows: coalesced b128 global->LDS, 8 WMMAs (16 rows each,
// K=0..2 live), raw D columns 0..2 -> LDS transpose, then the fp16 roundtrip
// + rescale + bias run in the full-wave epilogue (all 32 lanes useful) fused
// with coalesced non-temporal b128 stores. 256 thr = 8 waves = 1024 rows/blk.
//
// qi, qw are integers with |q| <= 127: exact in f16; dot products <= 48387:
// exact in the f32 WMMA accumulator -> WMMA path is faithful to the
// reference integer GEMM.
// ---------------------------------------------------------------------------
__global__ void __launch_bounds__(256)
qlin_wmma(const float* __restrict__ x, const float* __restrict__ w,
          const float* __restrict__ bias, const unsigned* __restrict__ ws,
          float* __restrict__ out) {
    __shared__ float lin [8][384];   // per-wave: 128 rows x 3
    __shared__ float lout[8][384];   // raw accumulators, transposed layout

    const int  tid  = threadIdx.x;
    const int  wv   = tid >> 5;      // wave id in block (wave32)
    const int  lane = tid & 31;
    const long long rowBase = ((long long)blockIdx.x * 8 + wv) * 128;

    // scales (match reference: max/127, 0 -> 1.0); one divide each, hoisted
    float i_scale = __uint_as_float(ws[0]) / 127.0f;
    if (i_scale == 0.0f) i_scale = 1.0f;
    float w_scale = __uint_as_float(ws[1]) / 127.0f;
    if (w_scale == 0.0f) w_scale = 1.0f;
    const float inv_i = 1.0f / i_scale;
    const float inv_w = 1.0f / w_scale;
    const float os    = w_scale * i_scale;   // output rescale factor

    // ---- stage 128 rows (384 floats) into LDS, fully coalesced b128 ----
    const f32x4* src = (const f32x4*)(x + rowBase * 3);
    f32x4*       dst = (f32x4*)(&lin[wv][0]);
#pragma unroll
    for (int i = 0; i < 3; ++i) dst[i * 32 + lane] = src[i * 32 + lane];

    // ---- B matrix: qw.T (3x3) padded into 32x16 f16.
    // B[k][n] lives in element k of lane n (k<8, lanes 0-15); rest zero.
    v16h B = {};
    if (lane < 3) {
        B[0] = (_Float16)rintf(w[lane * 3 + 0] * inv_w);
        B[1] = (_Float16)rintf(w[lane * 3 + 1] * inv_w);
        B[2] = (_Float16)rintf(w[lane * 3 + 2] * inv_w);
    }

    // per-lane rotated bias registers for the epilogue:
    // element c of vector i maps to column n = (lane + 2*i + c) mod 3
    const float b0 = bias[0], b1 = bias[1], b2 = bias[2];
    const int rmod = lane % 3;
    const float br0 = (rmod == 0) ? b0 : ((rmod == 1) ? b1 : b2);  // (rmod+0)%3
    const float br1 = (rmod == 0) ? b1 : ((rmod == 1) ? b2 : b0);  // (rmod+1)%3
    const float br2 = (rmod == 0) ? b2 : ((rmod == 1) ? b0 : b1);  // (rmod+2)%3
    const float brot[3] = {br0, br1, br2};

    const int  n      = lane & 15;   // output column this lane holds in D
    const int  hi     = lane >> 4;   // 0: M=0..7, 1: M=8..15
    const bool writer = (n < 3);

    __syncthreads();

    // A hoisted: lanes 16-31 and elements 3..15 stay zero across the loop
    v16h A = {};
#pragma unroll
    for (int j = 0; j < 8; ++j) {
        // A: 16 rows of quantized x in K=0..2 (elements 0..2, lanes 0..15)
        if (lane < 16) {
            const int r = j * 16 + lane;
            A[0] = (_Float16)rintf(lin[wv][3 * r + 0] * inv_i);
            A[1] = (_Float16)rintf(lin[wv][3 * r + 1] * inv_i);
            A[2] = (_Float16)rintf(lin[wv][3 * r + 2] * inv_i);
        }
        v8f C = {};
        v8f D = __builtin_amdgcn_wmma_f32_16x16x32_f16(
            /*neg_a=*/false, A, /*neg_b=*/false, B,
            /*c_mod=*/(short)0, C, /*reuse_a=*/false, /*reuse_b=*/false);

        if (writer) {
            // store raw accumulators; math deferred to full-wave epilogue
#pragma unroll
            for (int r8 = 0; r8 < 8; ++r8) {
                const int m = j * 16 + hi * 8 + r8;       // row within wave tile
                lout[wv][m * 3 + n] = D[r8];
            }
        }
    }

    __syncthreads();

    // ---- full-wave epilogue: fp16 roundtrip + fma(os, bias), NT b128 out ----
    const f32x4* osrc = (const f32x4*)(&lout[wv][0]);
    f32x4*       gdst = (f32x4*)(out + rowBase * 3);
#pragma unroll
    for (int i = 0; i < 3; ++i) {
        f32x4 v = osrc[i * 32 + lane];
        f32x4 o;
        o.x = fmaf((float)(_Float16)v.x, os, brot[(2 * i + 0) % 3]);
        o.y = fmaf((float)(_Float16)v.y, os, brot[(2 * i + 1) % 3]);
        o.z = fmaf((float)(_Float16)v.z, os, brot[(2 * i + 2) % 3]);
        o.w = fmaf((float)(_Float16)v.w, os, brot[(2 * i + 3) % 3]);
        __builtin_nontemporal_store(o, &gdst[i * 32 + lane]);
    }
}

// ---------------------------------------------------------------------------
extern "C" void kernel_launch(void* const* d_in, const int* in_sizes, int n_in,
                              void* d_out, int out_size, void* d_ws, size_t ws_size,
                              hipStream_t stream) {
    const float* x    = (const float*)d_in[0];   // [4194304, 3]
    const float* w    = (const float*)d_in[1];   // [3, 3]
    const float* bias = (const float*)d_in[2];   // [3]
    float*       out  = (float*)d_out;
    unsigned*    ws   = (unsigned*)d_ws;

    const long long totalFloats = (long long)in_sizes[0];
    const long long nrows       = totalFloats / 3;

    qlin_ws_init<<<1, 32, 0, stream>>>(ws);

    const int n4 = (int)(totalFloats / 4);
    qlin_absmax<<<1024, 256, 0, stream>>>((const f32x4*)x, n4, w, ws);

    const int grid = (int)(nrows / 1024);        // 4096 blocks of 1024 rows
    qlin_wmma<<<grid, 256, 0, stream>>>(x, w, bias, ws, out);
}
